// GAT_38714835206187
// MI455X (gfx1250) — matrix operand order repaired and checked by reading, not compiled, not run on hardware
//
#include <hip/hip_runtime.h>
#include <math.h>

// ---------------------------------------------------------------------------
// GAT forward for MI455X (gfx1250): WMMA bf16 GEMMs + L2-atomic edge softmax.
// ---------------------------------------------------------------------------

typedef __attribute__((ext_vector_type(16))) __bf16 v16bf;
typedef __attribute__((ext_vector_type(8)))  float  v8f;

#define NEG_SLOPE 0.2f

__device__ __forceinline__ unsigned short f2bf_bits(float f) {
  unsigned u = __float_as_uint(f);
  u = (u + 0x7FFFu + ((u >> 16) & 1u)) >> 16;   // round-to-nearest-even
  return (unsigned short)u;
}
__device__ __forceinline__ __bf16 f2bf(float f) {
  unsigned short s = f2bf_bits(f);
  return __builtin_bit_cast(__bf16, s);
}

// float atomic max via monotonic integer ordering (init with -1e30f)
__device__ __forceinline__ void atomicMaxF(float* addr, float v) {
  if (v >= 0.0f) atomicMax((int*)addr, __float_as_int(v));
  else           atomicMin((unsigned int*)addr, __float_as_uint(v));
}

// ---------------------------------------------------------------------------
// Pack W[64][64] (row-major, f32) into WMMA B-fragment lane order (bf16).
// Fragment q = kc*4 + nt (kc: K-half of 32, nt: 16-col tile). For lane L:
// col = L&15, klo = (L>>4)*16, element j corresponds to K = kc*32 + klo + j.
// wp[((q*32)+L)*16 + j] = bf16( W[(kc*32+klo+j)*64 + nt*16 + col] )
// ---------------------------------------------------------------------------
__global__ void k_pack_w(const float* __restrict__ W, unsigned short* __restrict__ wp) {
  int t = blockIdx.x * blockDim.x + threadIdx.x;
  if (t >= 64 * 64) return;
  int j    = t & 15;
  int lane = (t >> 4) & 31;
  int q    = t >> 9;            // 0..7
  int kc   = q >> 2, nt = q & 3;
  int col  = lane & 15;
  int klo  = (lane >> 4) * 16;
  int K    = kc * 32 + klo + j;
  wp[t] = f2bf_bits(W[K * 64 + nt * 16 + col]);
}

// ---------------------------------------------------------------------------
// C[N,64] = A[N,64] @ W[64,64] via v_wmma_f32_16x16x32_bf16.
// One wave per 16-row tile; 4 column tiles x 2 K-steps = 8 WMMAs per wave.
// A converted f32->bf16 on the fly (two contiguous 8-float segments/lane).
// ---------------------------------------------------------------------------
__global__ void k_gemm_n64(const float* __restrict__ A,
                           const unsigned short* __restrict__ wp,
                           float* __restrict__ C, int nrows) {
  int wave   = threadIdx.x >> 5;
  int lane   = threadIdx.x & 31;
  int tile   = blockIdx.x * (blockDim.x >> 5) + wave;
  int ntiles = (nrows + 15) >> 4;
  if (tile >= ntiles) return;                 // wave-uniform: EXEC stays full
  int row0 = tile << 4;
  int half = lane >> 4;                       // 0/1
  int lrow = row0 + (lane & 15);
  if (lrow >= nrows) lrow = nrows - 1;        // clamp loads on ragged tail
  const float* arow = A + (size_t)lrow * 64;
  const v16bf* wv = (const v16bf*)wp;

  v8f acc0 = {}, acc1 = {}, acc2 = {}, acc3 = {};
#pragma unroll
  for (int kc = 0; kc < 2; ++kc) {
    int base = kc * 32 + half * 16;
    v16bf a;
#pragma unroll
    for (int j = 0; j < 8; ++j) a[j]     = f2bf(arow[base + j]);
#pragma unroll
    for (int j = 0; j < 8; ++j) a[8 + j] = f2bf(arow[base + 8 + j]);
    v16bf b0 = wv[(kc * 4 + 0) * 32 + lane];
    v16bf b1 = wv[(kc * 4 + 1) * 32 + lane];
    v16bf b2 = wv[(kc * 4 + 2) * 32 + lane];
    v16bf b3 = wv[(kc * 4 + 3) * 32 + lane];
    acc0 = __builtin_amdgcn_wmma_f32_16x16x32_bf16(false, a, false, b0, (short)0, acc0, false, false);
    acc1 = __builtin_amdgcn_wmma_f32_16x16x32_bf16(false, a, false, b1, (short)0, acc1, false, false);
    acc2 = __builtin_amdgcn_wmma_f32_16x16x32_bf16(false, a, false, b2, (short)0, acc2, false, false);
    acc3 = __builtin_amdgcn_wmma_f32_16x16x32_bf16(false, a, false, b3, (short)0, acc3, false, false);
  }
  // D layout: VGPR r -> row M = r + half*8, col N = lane&15
  int col = lane & 15;
  int mb  = half * 8;
#pragma unroll
  for (int r = 0; r < 8; ++r) {
    int row = row0 + mb + r;
    if (row < nrows) {
      float* crow = C + (size_t)row * 64 + col;
      crow[0]  = acc0[r];
      crow[16] = acc1[r];
      crow[32] = acc2[r];
      crow[48] = acc3[r];
    }
  }
}

// hs[i] = h[i,:].a_src ; hd[i] = h[i,:].a_dst
__global__ void k_dots(const float* __restrict__ h, const float* __restrict__ as,
                       const float* __restrict__ ad, float* __restrict__ hs,
                       float* __restrict__ hd, int n) {
  int i = blockIdx.x * blockDim.x + threadIdx.x;
  if (i >= n) return;
  const float4* row = (const float4*)(h + (size_t)i * 64);
  const float4* a4  = (const float4*)as;
  const float4* d4  = (const float4*)ad;
  float ss = 0.f, sd = 0.f;
#pragma unroll
  for (int q = 0; q < 16; ++q) {
    float4 r = row[q], a = a4[q], d = d4[q];
    ss += r.x * a.x + r.y * a.y + r.z * a.z + r.w * a.w;
    sd += r.x * d.x + r.y * d.y + r.z * d.z + r.w * d.w;
  }
  hs[i] = ss;
  hd[i] = sd;
}

__global__ void k_init_node(float* __restrict__ agg, float* __restrict__ maxl,
                            float* __restrict__ denom, int n) {
  int t = blockIdx.x * blockDim.x + threadIdx.x;
  if (t < n * 64) agg[t] = 0.f;
  if (t < n) { maxl[t] = -1e30f; denom[t] = 0.f; }
}

__global__ void k_init_pool(float* pmax1, float* psum1, float* pmax2, float* psum2,
                            float* cnt, int g) {
  int t = blockIdx.x * blockDim.x + threadIdx.x;
  if (t < g * 64) { pmax1[t] = -1e30f; psum1[t] = 0.f; pmax2[t] = -1e30f; psum2[t] = 0.f; }
  if (t < g) cnt[t] = 0.f;
}

// logits + segment-max over dst
__global__ void k_edge_logits(const int* __restrict__ ei, int E, int etot,
                              const float* __restrict__ hs, const float* __restrict__ hd,
                              float* __restrict__ logits, float* __restrict__ maxl) {
  int e = blockIdx.x * blockDim.x + threadIdx.x;
  if (e >= etot) return;
  int s, d;
  if (e < E) { s = ei[e]; d = ei[E + e]; } else { s = d = e - E; }
  float l = hs[s] + hd[d];
  l = (l > 0.f) ? l : NEG_SLOPE * l;
  logits[e] = l;
  atomicMaxF(&maxl[d], l);
}

// exp(logit - max[dst]) + segment-sum denom; weight stored in place of logit
__global__ void k_edge_exp(const int* __restrict__ ei, int E, int etot,
                           const float* __restrict__ maxl,
                           float* __restrict__ logits, float* __restrict__ denom) {
  int e = blockIdx.x * blockDim.x + threadIdx.x;
  if (e >= etot) return;
  int d = (e < E) ? ei[E + e] : (e - E);
  float w = __expf(logits[e] - maxl[d]);
  logits[e] = w;
  atomicAdd(&denom[d], w);
}

// wave-per-edge: 32 lanes cover 64 features (float2 each); coalesced gather of
// h[src] row from L2 + coalesced f32 atomic scatter into agg[dst] row.
__global__ void k_edge_agg(const int* __restrict__ ei, int E, int etot,
                           const float* __restrict__ w, const float* __restrict__ denom,
                           const float* __restrict__ h, float* __restrict__ agg) {
  int t    = blockIdx.x * blockDim.x + threadIdx.x;
  int e    = t >> 5;
  int lane = t & 31;
  if (e >= etot) return;
  int s, d;
  if (e < E) { s = ei[e]; d = ei[E + e]; } else { s = d = e - E; }
  float alpha = w[e] / fmaxf(denom[d], 1e-16f);
  float2 hv = *(const float2*)(h + (size_t)s * 64 + lane * 2);
  float* out = agg + (size_t)d * 64 + lane * 2;
  atomicAdd(out,     alpha * hv.x);
  atomicAdd(out + 1, alpha * hv.y);
}

// h = relu(agg + b); pool (max, sum, count) per graph via L2 atomics
__global__ void k_bias_relu_pool(float* __restrict__ hio, const float* __restrict__ bias,
                                 const int* __restrict__ batch,
                                 float* __restrict__ pmax, float* __restrict__ psum,
                                 float* __restrict__ cnt, int n, int do_cnt) {
  int t = blockIdx.x * blockDim.x + threadIdx.x;
  if (t >= n * 64) return;
  int i = t >> 6, f = t & 63;
  float v = hio[t] + bias[f];
  v = fmaxf(v, 0.f);
  hio[t] = v;
  int g = batch[i];
  atomicMaxF(&pmax[g * 64 + f], v);
  atomicAdd(&psum[g * 64 + f], v);
  if (do_cnt && f == 0) atomicAdd(&cnt[g], 1.f);
}

// comb[g,k] = x1[g,k] + x2[g,k]  (x = [max | mean]); also init out[g] = lin2_b
__global__ void k_comb(const float* pmax1, const float* psum1, const float* pmax2,
                       const float* psum2, const float* cnt, const float* lin2_b,
                       float* __restrict__ comb, float* __restrict__ out, int g) {
  int t = blockIdx.x * blockDim.x + threadIdx.x;
  if (t >= g * 128) return;
  int gi = t >> 7, k = t & 127;
  float v;
  if (k < 64) {
    v = pmax1[gi * 64 + k] + pmax2[gi * 64 + k];
  } else {
    float c = fmaxf(cnt[gi], 1.f);
    v = (psum1[gi * 64 + (k - 64)] + psum2[gi * 64 + (k - 64)]) / c;
  }
  comb[t] = v;
  if (t < g) out[t] = lin2_b[0];
}

// thread per (g, j): z = relu(comb[g,:] @ lin1_W[:,j] + b); out[g] += z*lin2_W[j]
__global__ void k_mlp(const float* __restrict__ comb, const float* __restrict__ w1,
                      const float* __restrict__ b1v, const float* __restrict__ w2,
                      float* __restrict__ out, int g) {
  int t = blockIdx.x * blockDim.x + threadIdx.x;
  if (t >= g * 64) return;
  int gi = t >> 6, j = t & 63;
  const float* c = comb + gi * 128;
  float a = b1v[j];
#pragma unroll 8
  for (int k = 0; k < 128; ++k) a += c[k] * w1[k * 64 + j];
  a = fmaxf(a, 0.f);
  atomicAdd(&out[gi], a * w2[j]);
}

// ---------------------------------------------------------------------------
extern "C" void kernel_launch(void* const* d_in, const int* in_sizes, int n_in,
                              void* d_out, int out_size, void* d_ws, size_t ws_size,
                              hipStream_t stream) {
  const float* x     = (const float*)d_in[0];
  const int*   ei    = (const int*)d_in[1];
  const int*   batch = (const int*)d_in[2];
  const float* W1    = (const float*)d_in[3];
  const float* as1   = (const float*)d_in[4];
  const float* ad1   = (const float*)d_in[5];
  const float* b1    = (const float*)d_in[6];
  const float* W2    = (const float*)d_in[7];
  const float* as2   = (const float*)d_in[8];
  const float* ad2   = (const float*)d_in[9];
  const float* b2    = (const float*)d_in[10];
  const float* lin1W = (const float*)d_in[11];
  const float* lin1b = (const float*)d_in[12];
  const float* lin2W = (const float*)d_in[13];
  const float* lin2b = (const float*)d_in[14];
  float* out = (float*)d_out;

  const int N    = in_sizes[0] / 64;
  const int E    = in_sizes[1] / 2;
  const int G    = out_size;
  const int Etot = E + N;
  const size_t N64 = (size_t)N * 64;

  // workspace layout
  char* wsb = (char*)d_ws;
  unsigned short* wpack = (unsigned short*)wsb;       // 4096 bf16 = 8 KB
  float* fb = (float*)(wsb + 8192);
  float* bufA   = fb; fb += N64;                      // GEMM output h
  float* bufB   = fb; fb += N64;                      // agg target / layer input
  float* logits = fb; fb += (size_t)Etot;             // logits -> weights
  float* hs     = fb; fb += N;
  float* hd     = fb; fb += N;
  float* maxl   = fb; fb += N;
  float* denom  = fb; fb += N;
  float* pmax1  = fb; fb += (size_t)G * 64;
  float* psum1  = fb; fb += (size_t)G * 64;
  float* pmax2  = fb; fb += (size_t)G * 64;
  float* psum2  = fb; fb += (size_t)G * 64;
  float* cnt    = fb; fb += G;
  float* comb   = fb; fb += (size_t)G * 128;

  const int TB = 256;
  const int ntiles  = (N + 15) / 16;
  const int gGemm   = (ntiles + 7) / 8;               // 8 waves per block
  const int gNode   = (N + TB - 1) / TB;
  const int gNF     = (int)((N64 + TB - 1) / TB);
  const int gEdge   = (Etot + TB - 1) / TB;
  const int gAgg    = (int)(((size_t)Etot * 32 + TB - 1) / TB);

  k_init_pool<<<(G * 64 + TB - 1) / TB, TB, 0, stream>>>(pmax1, psum1, pmax2, psum2, cnt, G);

  // ---- layer 1 ----
  k_pack_w<<<16, TB, 0, stream>>>(W1, wpack);
  k_gemm_n64<<<gGemm, TB, 0, stream>>>(x, wpack, bufA, N);
  k_dots<<<gNode, TB, 0, stream>>>(bufA, as1, ad1, hs, hd, N);
  k_init_node<<<gNF, TB, 0, stream>>>(bufB, maxl, denom, N);
  k_edge_logits<<<gEdge, TB, 0, stream>>>(ei, E, Etot, hs, hd, logits, maxl);
  k_edge_exp<<<gEdge, TB, 0, stream>>>(ei, E, Etot, maxl, logits, denom);
  k_edge_agg<<<gAgg, TB, 0, stream>>>(ei, E, Etot, logits, denom, bufA, bufB);
  k_bias_relu_pool<<<gNF, TB, 0, stream>>>(bufB, b1, batch, pmax1, psum1, cnt, N, 1);

  // ---- layer 2 (h1 lives in bufB; GEMM consumes it, then bufB becomes agg) ----
  k_pack_w<<<16, TB, 0, stream>>>(W2, wpack);
  k_gemm_n64<<<gGemm, TB, 0, stream>>>(bufB, wpack, bufA, N);
  k_dots<<<gNode, TB, 0, stream>>>(bufA, as2, ad2, hs, hd, N);
  k_init_node<<<gNF, TB, 0, stream>>>(bufB, maxl, denom, N);
  k_edge_logits<<<gEdge, TB, 0, stream>>>(ei, E, Etot, hs, hd, logits, maxl);
  k_edge_exp<<<gEdge, TB, 0, stream>>>(ei, E, Etot, maxl, logits, denom);
  k_edge_agg<<<gAgg, TB, 0, stream>>>(ei, E, Etot, logits, denom, bufA, bufB);
  k_bias_relu_pool<<<gNF, TB, 0, stream>>>(bufB, b2, batch, pmax2, psum2, cnt, N, 0);

  // ---- readout head ----
  k_comb<<<(G * 128 + TB - 1) / TB, TB, 0, stream>>>(pmax1, psum1, pmax2, psum2, cnt,
                                                     lin2b, comb, out, G);
  k_mlp<<<(G * 64 + TB - 1) / TB, TB, 0, stream>>>(comb, lin1W, lin1b, lin2W, out, G);
}
